// AgentCentricEncoder_62122406969475
// MI455X (gfx1250) — compile-verified, matrix operand order checked by reading, not run on hardware
//
#include <hip/hip_runtime.h>

#define BB 32
#define NA 128
#define NM 512
#define DD 256
#define HH 8
#define LL 2
#define TK 32
#define FF 1024
#define DH 32

typedef __bf16 bf16_t;
typedef __attribute__((ext_vector_type(16))) bf16_t v16bf;
typedef __attribute__((ext_vector_type(8)))  bf16_t v8bf;
typedef __attribute__((ext_vector_type(8)))  float  v8f;

__device__ __forceinline__ unsigned short f2bf(float f) {
    union { float f; unsigned u; } v; v.f = f;
    unsigned r = v.u + 0x7fffu + ((v.u >> 16) & 1u); // round-to-nearest-even
    return (unsigned short)(r >> 16);
}

__device__ __forceinline__ v16bf make_frag(const unsigned short* lo, const unsigned short* hi) {
    v8bf l = *reinterpret_cast<const v8bf*>(lo);
    v8bf h = *reinterpret_cast<const v8bf*>(hi);
    v16bf r;
#pragma unroll
    for (int i = 0; i < 8; ++i) { r[i] = l[i]; r[i + 8] = h[i]; }
    return r;
}

// ---------------- transpose + convert: src [cnt][K][N] f32 -> dst [cnt][N][K] bf16
__global__ void __launch_bounds__(256) transpose_bf16_kernel(
    const float* __restrict__ src, unsigned short* __restrict__ dst,
    long total, int Kd, int Nd) {
    long i = (long)blockIdx.x * 256 + threadIdx.x;
    if (i >= total) return;
    long per = (long)Kd * Nd;
    long m = i / per, r = i % per;
    int k = (int)(r / Nd), n = (int)(r % Nd);
    dst[m * per + (long)n * Kd + k] = f2bf(src[i]);
}

// ---------------- cast f32 -> bf16
__global__ void __launch_bounds__(256) cast_bf16_kernel(
    const float* __restrict__ in, unsigned short* __restrict__ out, long n) {
    long i = (long)blockIdx.x * 256 + threadIdx.x;
    if (i < n) out[i] = f2bf(in[i]);
}

// ---------------- WMMA GEMM: C[M,N] = A_bf16[M,K] * Bt_bf16[N,K]^T + bias, opt relu, opt bf16 out
// Per-wave tile: 32 rows x 64 cols -> 8 WMMAs per K=32 step, B-fragments reused 2x.
__global__ void __launch_bounds__(256) gemm_bf16_kernel(
    const unsigned short* __restrict__ A, const unsigned short* __restrict__ Bt,
    const float* __restrict__ bias, float* __restrict__ Cf,
    unsigned short* __restrict__ Cb, int Md, int Nd, int Kd, int relu) {
    int lane = threadIdx.x & 31;
    int wave = blockIdx.x * 8 + (threadIdx.x >> 5);
    int nstrips = Nd >> 6;               // 64-wide col strips
    int mstrips = Md >> 5;               // 32-high row strips
    int total = mstrips * nstrips;
    if (wave >= total) return;           // wave-uniform: EXEC stays all-1s for WMMA
    int mt = wave / nstrips, ns = wave % nstrips;
    int r = lane & 15, hh = lane >> 4;
    const unsigned short* arow0 = A + (size_t)(mt * 32 +  0 + r) * Kd;
    const unsigned short* arow1 = A + (size_t)(mt * 32 + 16 + r) * Kd;
    const unsigned short* brow0 = Bt + (size_t)(ns * 64 +  0 + r) * Kd;
    const unsigned short* brow1 = Bt + (size_t)(ns * 64 + 16 + r) * Kd;
    const unsigned short* brow2 = Bt + (size_t)(ns * 64 + 32 + r) * Kd;
    const unsigned short* brow3 = Bt + (size_t)(ns * 64 + 48 + r) * Kd;
    v8f acc[2][4];
#pragma unroll
    for (int m = 0; m < 2; ++m)
#pragma unroll
        for (int s = 0; s < 4; ++s) acc[m][s] = (v8f){};
    for (int k0 = 0; k0 < Kd; k0 += 32) {
        int klo = k0 + hh * 8, khi = k0 + 16 + hh * 8;
        if (k0 + 32 < Kd) {
            __builtin_prefetch(arow0 + klo + 32, 0, 3);
            __builtin_prefetch(arow1 + klo + 32, 0, 3);
        }
        v16bf a0 = make_frag(arow0 + klo, arow0 + khi);
        v16bf a1 = make_frag(arow1 + klo, arow1 + khi);
        v16bf b0 = make_frag(brow0 + klo, brow0 + khi);
        v16bf b1 = make_frag(brow1 + klo, brow1 + khi);
        v16bf b2 = make_frag(brow2 + klo, brow2 + khi);
        v16bf b3 = make_frag(brow3 + klo, brow3 + khi);
        acc[0][0] = __builtin_amdgcn_wmma_f32_16x16x32_bf16(false, a0, false, b0, (short)0, acc[0][0], false, false);
        acc[1][0] = __builtin_amdgcn_wmma_f32_16x16x32_bf16(false, a1, false, b0, (short)0, acc[1][0], false, false);
        acc[0][1] = __builtin_amdgcn_wmma_f32_16x16x32_bf16(false, a0, false, b1, (short)0, acc[0][1], false, false);
        acc[1][1] = __builtin_amdgcn_wmma_f32_16x16x32_bf16(false, a1, false, b1, (short)0, acc[1][1], false, false);
        acc[0][2] = __builtin_amdgcn_wmma_f32_16x16x32_bf16(false, a0, false, b2, (short)0, acc[0][2], false, false);
        acc[1][2] = __builtin_amdgcn_wmma_f32_16x16x32_bf16(false, a1, false, b2, (short)0, acc[1][2], false, false);
        acc[0][3] = __builtin_amdgcn_wmma_f32_16x16x32_bf16(false, a0, false, b3, (short)0, acc[0][3], false, false);
        acc[1][3] = __builtin_amdgcn_wmma_f32_16x16x32_bf16(false, a1, false, b3, (short)0, acc[1][3], false, false);
    }
#pragma unroll
    for (int m = 0; m < 2; ++m) {
        int rowb = mt * 32 + m * 16 + hh * 8;
#pragma unroll
        for (int s = 0; s < 4; ++s) {
            int col = ns * 64 + s * 16 + r;
            float bsv = bias[col];
#pragma unroll
            for (int i = 0; i < 8; ++i) {
                float vv = acc[m][s][i] + bsv;
                if (relu) vv = fmaxf(vv, 0.0f);
                size_t o = (size_t)(rowb + i) * Nd + col;
                if (Cb) Cb[o] = f2bf(vv); else Cf[o] = vv;
            }
        }
    }
}

// ---------------- top-K neighbor selection (stable, ascending distance)
__global__ void __launch_bounds__(256) topk_kernel(
    const float* __restrict__ qpos, const float* __restrict__ kpos,
    const unsigned char* __restrict__ kmask, int* __restrict__ idx,
    unsigned char* __restrict__ valid, int Nq, int Nk) {
    int t = blockIdx.x * 256 + threadIdx.x;
    if (t >= BB * Nq) return;
    int b = t / Nq;
    float qx = qpos[(size_t)t * 2], qy = qpos[(size_t)t * 2 + 1];
    const float* kp = kpos + (size_t)b * Nk * 2;
    const unsigned char* km = kmask + (size_t)b * Nk;
    float dist[TK]; int di[TK];
#pragma unroll
    for (int i = 0; i < TK; ++i) { dist[i] = 3.4e38f; di[i] = 0; }
    for (int j = 0; j < Nk; ++j) {
        if (!km[j]) continue;
        float dx = qx - kp[2 * j], dy = qy - kp[2 * j + 1];
        float dd = sqrtf(dx * dx + dy * dy);
        if (dd < dist[TK - 1]) {
            int p = TK - 1;
            while (p > 0 && dist[p - 1] > dd) { dist[p] = dist[p - 1]; di[p] = di[p - 1]; --p; }
            dist[p] = dd; di[p] = j;
        }
    }
    for (int k = 0; k < TK; ++k) {
        idx[(size_t)t * TK + k] = di[k];
        valid[(size_t)t * TK + k] = (dist[k] < 3.0e38f) ? 1 : 0;
    }
}

// ---------------- RoPE in-place on [B,N,H,32]
__global__ void __launch_bounds__(256) rope_kernel(
    float* __restrict__ x, const float* __restrict__ pos, int N) {
    int t = blockIdx.x * 256 + threadIdx.x;
    int total = BB * N * HH * 16;
    if (t >= total) return;
    int j = t & 15, h = (t >> 4) & 7, bn = t >> 7;
    float p = (j < 8) ? pos[(size_t)bn * 2] : pos[(size_t)bn * 2 + 1];
    int jj = j & 7;
    float inv = __expf(-(float)jj * 1.1512925465f); // ln(10000)/8
    float ang = p * inv, s, c;
    __sincosf(ang, &s, &c);
    size_t base = ((size_t)bn * HH + h) * DH;
    float x1 = x[base + j], x2 = x[base + 16 + j];
    x[base + j]      = x1 * c - x2 * s;
    x[base + 16 + j] = x1 * s + x2 * c;
}

// ---------------- gathered sparse attention: one wave32 per (b,q,h)
__global__ void __launch_bounds__(256) attn_kernel(
    const float* __restrict__ Q, const float* __restrict__ Kf,
    const float* __restrict__ V, const int* __restrict__ idx,
    const unsigned char* __restrict__ valid, float* __restrict__ O,
    int Nq, int Nk) {
    __shared__ float sw[8][TK];
    __shared__ int   si[8][TK];
    int lane = threadIdx.x & 31, wv = threadIdx.x >> 5;
    long gw = (long)blockIdx.x * 8 + wv;
    long total = (long)BB * Nq * HH;
    bool act = gw < total;
    int b = 0, q = 0, h = 0; float w = 0.0f; int id = 0;
    if (act) {
        h = (int)(gw % HH); long t = gw / HH; q = (int)(t % Nq); b = (int)(t / Nq);
        size_t qi = ((size_t)b * Nq + q) * TK + lane;
        id = idx[qi];
        unsigned char vl = valid[qi];
        const float* qp = Q + (((size_t)b * Nq + q) * HH + h) * DH;
        const float* kp = Kf + (((size_t)b * Nk + id) * HH + h) * DH;
        float s = 0.0f;
#pragma unroll
        for (int d = 0; d < DH; ++d) s += qp[d] * kp[d];
        s *= 0.17677669529663687f; // 1/sqrt(32)
        if (!vl) s = -1.0e9f;
        float m = s;
        for (int o = 16; o > 0; o >>= 1) m = fmaxf(m, __shfl_xor(m, o, 32));
        float e = __expf(s - m);
        float sum = e;
        for (int o = 16; o > 0; o >>= 1) sum += __shfl_xor(sum, o, 32);
        w = e / sum;
    }
    sw[wv][lane] = w; si[wv][lane] = id;
    __syncthreads();
    if (act) {
        float o = 0.0f;
        for (int k = 0; k < TK; ++k) {
            int ik = si[wv][k];
            o += sw[wv][k] * V[(((size_t)b * Nk + ik) * HH + h) * DH + lane];
        }
        O[(((size_t)b * Nq + q) * HH + h) * DH + lane] = o;
    }
}

// ---------------- residual + LayerNorm (+optional mask zero): one wave per row, D=256
__global__ void __launch_bounds__(256) ln_kernel(
    const float* __restrict__ base, const float* __restrict__ add,
    float* __restrict__ out, const float* __restrict__ g,
    const float* __restrict__ be, const unsigned char* __restrict__ mask, int rows) {
    int lane = threadIdx.x & 31;
    int row = (blockIdx.x * 256 + threadIdx.x) >> 5;
    if (row >= rows) return;
    const float* b0 = base + (size_t)row * DD;
    const float* a0 = add  + (size_t)row * DD;
    float v[8]; float s = 0.0f;
#pragma unroll
    for (int i = 0; i < 8; ++i) { v[i] = b0[lane + 32 * i] + a0[lane + 32 * i]; s += v[i]; }
    for (int o = 16; o > 0; o >>= 1) s += __shfl_xor(s, o, 32);
    float mean = s * (1.0f / DD);
    float vs = 0.0f;
#pragma unroll
    for (int i = 0; i < 8; ++i) { float d = v[i] - mean; vs += d * d; }
    for (int o = 16; o > 0; o >>= 1) vs += __shfl_xor(vs, o, 32);
    float inv = rsqrtf(vs * (1.0f / DD) + 1e-5f);
    float z = (mask && !mask[row]) ? 0.0f : 1.0f;
#pragma unroll
    for (int i = 0; i < 8; ++i) {
        int c = lane + 32 * i;
        out[(size_t)row * DD + c] = z * (((v[i] - mean) * inv) * g[c] + be[c]);
    }
}

extern "C" void kernel_launch(void* const* d_in, const int* in_sizes, int n_in,
                              void* d_out, int out_size, void* d_ws, size_t ws_size,
                              hipStream_t stream) {
    (void)in_sizes; (void)n_in; (void)out_size; (void)ws_size;
    const float* agent_feat = (const float*)d_in[0];
    const float* map_feat   = (const float*)d_in[1];
    const float* agent_pos  = (const float*)d_in[2];
    const float* map_pos    = (const float*)d_in[3];
    const float* wq = (const float*)d_in[4];
    const float* wk = (const float*)d_in[5];
    const float* wv = (const float*)d_in[6];
    const float* wo = (const float*)d_in[7];
    const float* bq = (const float*)d_in[8];
    const float* bk = (const float*)d_in[9];
    const float* bv = (const float*)d_in[10];
    const float* bo = (const float*)d_in[11];
    const float* gamma = (const float*)d_in[12];
    const float* beta  = (const float*)d_in[13];
    const float* w1 = (const float*)d_in[14];
    const float* b1 = (const float*)d_in[15];
    const float* w2 = (const float*)d_in[16];
    const float* b2 = (const float*)d_in[17];
    const unsigned char* agent_mask = (const unsigned char*)d_in[18];
    const unsigned char* map_mask   = (const unsigned char*)d_in[19];

    // -------- workspace arena
    size_t off = 0; char* basep = (char*)d_ws;
    auto carve = [&](size_t bytes) -> void* {
        void* p = basep + off; off += (bytes + 255) & ~(size_t)255; return p;
    };
    const int RMAX = BB * NM; // 16384 rows max
    float* AF = (float*)carve((size_t)BB * NA * DD * 4);
    float* MF = (float*)carve((size_t)BB * NM * DD * 4);
    float* QB = (float*)carve((size_t)RMAX * DD * 4);
    float* KB = (float*)carve((size_t)RMAX * DD * 4);
    float* VB = (float*)carve((size_t)RMAX * DD * 4);
    float* OB = (float*)carve((size_t)RMAX * DD * 4);
    float* PB = (float*)carve((size_t)RMAX * DD * 4);
    unsigned short* XB = (unsigned short*)carve((size_t)RMAX * DD * 2);
    unsigned short* HB = (unsigned short*)carve((size_t)RMAX * FF * 2);
    int* IDX_MM = (int*)carve((size_t)BB * NM * TK * 4);
    unsigned char* VAL_MM = (unsigned char*)carve((size_t)BB * NM * TK);
    int* IDX_AA = (int*)carve((size_t)BB * NA * TK * 4);
    unsigned char* VAL_AA = (unsigned char*)carve((size_t)BB * NA * TK);
    int* IDX_AM = (int*)carve((size_t)BB * NA * TK * 4);
    unsigned char* VAL_AM = (unsigned char*)carve((size_t)BB * NA * TK);
    unsigned short* WQT = (unsigned short*)carve((size_t)LL * 3 * DD * DD * 2);
    unsigned short* WKT = (unsigned short*)carve((size_t)LL * 3 * DD * DD * 2);
    unsigned short* WVT = (unsigned short*)carve((size_t)LL * 3 * DD * DD * 2);
    unsigned short* WOT = (unsigned short*)carve((size_t)LL * 3 * DD * DD * 2);
    unsigned short* W1T = (unsigned short*)carve((size_t)LL * 3 * DD * FF * 2);
    unsigned short* W2T = (unsigned short*)carve((size_t)LL * 3 * DD * FF * 2);

    // -------- weight transpose+convert (once per call)
    {
        long ta = (long)LL * 3 * DD * DD;
        int ba = (int)((ta + 255) / 256);
        transpose_bf16_kernel<<<ba, 256, 0, stream>>>(wq, WQT, ta, DD, DD);
        transpose_bf16_kernel<<<ba, 256, 0, stream>>>(wk, WKT, ta, DD, DD);
        transpose_bf16_kernel<<<ba, 256, 0, stream>>>(wv, WVT, ta, DD, DD);
        transpose_bf16_kernel<<<ba, 256, 0, stream>>>(wo, WOT, ta, DD, DD);
        long tf = (long)LL * 3 * DD * FF;
        int bf = (int)((tf + 255) / 256);
        transpose_bf16_kernel<<<bf, 256, 0, stream>>>(w1, W1T, tf, DD, FF);
        transpose_bf16_kernel<<<bf, 256, 0, stream>>>(w2, W2T, tf, FF, DD);
    }

    // -------- feature init
    hipMemcpyAsync(AF, agent_feat, (size_t)BB * NA * DD * 4, hipMemcpyDeviceToDevice, stream);
    hipMemcpyAsync(MF, map_feat,   (size_t)BB * NM * DD * 4, hipMemcpyDeviceToDevice, stream);

    // -------- top-k (positions/masks are layer-invariant -> compute once)
    topk_kernel<<<(BB * NM + 255) / 256, 256, 0, stream>>>(map_pos,   map_pos, map_mask,   IDX_MM, VAL_MM, NM, NM);
    topk_kernel<<<(BB * NA + 255) / 256, 256, 0, stream>>>(agent_pos, agent_pos, agent_mask, IDX_AA, VAL_AA, NA, NA);
    topk_kernel<<<(BB * NA + 255) / 256, 256, 0, stream>>>(agent_pos, map_pos, map_mask,   IDX_AM, VAL_AM, NA, NM);

    auto launch_gemm = [&](const unsigned short* Aa, const unsigned short* Bt, const float* bias,
                           float* Cf, unsigned short* Cb, int Md, int Nd, int Kd, int relu) {
        int waves = (Md / 32) * (Nd / 64);
        gemm_bf16_kernel<<<(waves + 7) / 8, 256, 0, stream>>>(Aa, Bt, bias, Cf, Cb, Md, Nd, Kd, relu);
    };
    auto launch_cast = [&](const float* in, unsigned short* o, long n) {
        cast_bf16_kernel<<<(int)((n + 255) / 256), 256, 0, stream>>>(in, o, n);
    };

    auto block = [&](float* fq, const float* fk, const float* qp, const float* kp,
                     const int* idxp, const unsigned char* valp, int l, int t, int g0,
                     const unsigned char* mk, int Nq, int Nk) {
        int rq = BB * Nq, rk = BB * Nk;
        size_t wofs = (size_t)(l * 3 + t) * DD * DD;
        size_t bofs = (size_t)(l * 3 + t) * DD;
        launch_cast(fk, XB, (long)rk * DD);
        launch_gemm(XB, WKT + wofs, bk + bofs, KB, nullptr, rk, DD, DD, 0);
        launch_gemm(XB, WVT + wofs, bv + bofs, VB, nullptr, rk, DD, DD, 0);
        if (fq != fk) launch_cast(fq, XB, (long)rq * DD);
        launch_gemm(XB, WQT + wofs, bq + bofs, QB, nullptr, rq, DD, DD, 0);
        rope_kernel<<<(BB * Nq * HH * 16 + 255) / 256, 256, 0, stream>>>(QB, qp, Nq);
        rope_kernel<<<(BB * Nk * HH * 16 + 255) / 256, 256, 0, stream>>>(KB, kp, Nk);
        int aw = BB * Nq * HH;
        attn_kernel<<<(aw + 7) / 8, 256, 0, stream>>>(QB, KB, VB, idxp, valp, OB, Nq, Nk);
        launch_cast(OB, XB, (long)rq * DD);
        launch_gemm(XB, WOT + wofs, bo + bofs, PB, nullptr, rq, DD, DD, 0);
        ln_kernel<<<(rq + 7) / 8, 256, 0, stream>>>(fq, PB, fq,
            gamma + (size_t)(l * 6 + g0) * DD, beta + (size_t)(l * 6 + g0) * DD, nullptr, rq);
        // FFN
        launch_cast(fq, XB, (long)rq * DD);
        launch_gemm(XB, W1T + (size_t)(l * 3 + t) * DD * FF, b1 + (size_t)(l * 3 + t) * FF,
                    nullptr, HB, rq, FF, DD, 1);
        launch_gemm(HB, W2T + (size_t)(l * 3 + t) * DD * FF, b2 + bofs, PB, nullptr, rq, DD, FF, 0);
        ln_kernel<<<(rq + 7) / 8, 256, 0, stream>>>(fq, PB, fq,
            gamma + (size_t)(l * 6 + g0 + 1) * DD, beta + (size_t)(l * 6 + g0 + 1) * DD, mk, rq);
    };

    for (int l = 0; l < LL; ++l) {
        block(MF, MF, map_pos,   map_pos, IDX_MM, VAL_MM, l, 0, 0, map_mask,   NM, NM);
        block(AF, AF, agent_pos, agent_pos, IDX_AA, VAL_AA, l, 1, 2, agent_mask, NA, NA);
        block(AF, MF, agent_pos, map_pos, IDX_AM, VAL_AM, l, 2, 4, agent_mask, NA, NM);
    }

    hipMemcpyAsync(d_out, AF, (size_t)BB * NA * DD * 4, hipMemcpyDeviceToDevice, stream);
    hipMemcpyAsync((float*)d_out + (size_t)BB * NA * DD, MF,
                   (size_t)BB * NM * DD * 4, hipMemcpyDeviceToDevice, stream);
}
